// MambaClassifier_11708080849242
// MI455X (gfx1250) — matrix operand order repaired and checked by reading, not compiled, not run on hardware
//
#include <hip/hip_runtime.h>
#include <hip/hip_bf16.h>

// ---------------------------------------------------------------------------
// Mamba classifier for MI455X (gfx1250), wave32, bf16 WMMA + TDM path.
//   B=8 L=2048 D_MODEL=768 D_STATE=16 D_CONV=4 D_INNER=1536 DT_RANK=48 OUT=2
// Data path: f32 operands are converted to bf16 ONCE (precvt kernels), so the
// GEMM inner loops are pure 16-byte bf16 loads whose register image is the
// WMMA fragment (two v8bf loads + shufflevector), no converts in the K-loop.
// Pipeline:
//   P0 cvt       : x, in_proj_w, x_proj_w, dt_proj_w -> bf16 copies in ws
//   K1 gemm32x64 : xz[16384,3072] = x @ in_proj_w^T       (8 WMMA / k-step)
//   K2 conv      : xi_c = silu(causal conv(xz[:, :1536])), also bf16 copy
//   K3 gemm16x16 : dbc[16384,80]  = xi_c @ x_proj_w^T
//   K4 gemm TDM  : delta = softplus(dbc[:, :48] @ dt_proj_w^T + b)
//                  (bf16 B tile staged in LDS via tensor_load_to_lds)
//   K5 scan      : per (b,e) recurrence over L, fused z-gate + D-skip + mean
//   K6 fold      : M[2,1536] = fc_w @ out_proj_w (out_proj+pool+fc folded)
//   K7 final     : out[b,o] = sum_e ybar[b,e]*M[o,e] + fc_b[o]
// ---------------------------------------------------------------------------

#define BB       8
#define LL       2048
#define DMODEL   768
#define DSTATE   16
#define DINNER   1536
#define DTRANK   48
#define XPROJ_N  80           // DTRANK + 2*DSTATE
#define ROWS     (BB * LL)    // 16384

typedef __attribute__((ext_vector_type(16))) __bf16 v16bf;
typedef __attribute__((ext_vector_type(8)))  __bf16 v8bf;
typedef __attribute__((ext_vector_type(8)))  float  v8f;
typedef __attribute__((ext_vector_type(4)))  unsigned int u32x4;
typedef __attribute__((ext_vector_type(4)))  int   i32x4;
typedef __attribute__((ext_vector_type(8)))  int   i32x8;

__device__ __forceinline__ float sigmoid_f(float x) { return 1.f / (1.f + __expf(-x)); }
__device__ __forceinline__ float softplus_f(float x) {
  return (x > 20.f) ? x : __logf(1.f + __expf(x));
}

// ---- bf16 fragment loaders: a fragment is two 16-byte chunks ---------------
// (layouts per CDNA5 ISA §7.12.2; all chunk addresses 16B-aligned)
__device__ __forceinline__ v16bf frag16(const __bf16* p0, const __bf16* p1) {
  v8bf lo = *(const v8bf*)p0;
  v8bf hi = *(const v8bf*)p1;
  return __builtin_shufflevector(lo, hi, 0, 1, 2, 3, 4, 5, 6, 7,
                                         8, 9, 10, 11, 12, 13, 14, 15);
}
// A: 16x32. lane&15 = row M; lanes 0-15 K={0..7,16..23}, lanes 16-31 K={8..15,24..31}
__device__ __forceinline__ v16bf load_a_bf(const __bf16* rowp, int k0, int lane) {
  const int kb = (lane >> 4) << 3;  // 0 or 8
  return frag16(rowp + k0 + kb, rowp + k0 + 16 + kb);
}
// B: 32x16. lane&15 = col N; lanes 0-15 K={0..7,16..23}, lanes 16-31 K={8..15,24..31}
__device__ __forceinline__ v16bf load_b_bf(const __bf16* wrow, int k0, int lane) {
  const int kb = (lane >> 4) << 3;  // 0 or 8
  return frag16(wrow + k0 + kb, wrow + k0 + 16 + kb);
}
// guarded f32 chunk -> v8bf (chunk fully in- or out-of-bounds; kmax % 8 == 0)
__device__ __forceinline__ v8bf cvt_chunk8_guard(const float* rowp, int s, int kmax) {
  v8bf r;
  if (s < kmax) {
#pragma unroll
    for (int i = 0; i < 8; ++i) r[i] = (__bf16)rowp[s + i];
  } else {
#pragma unroll
    for (int i = 0; i < 8; ++i) r[i] = (__bf16)0.f;
  }
  return r;
}

// ---- Tensor Data Mover: 2D tile -> LDS (CDNA5 ISA) ---
// dsz_code: 0=1B 1=2B 2=4B 3=8B. Dims/strides in elements of that size.
__device__ __forceinline__ void tdm_load_2d(unsigned lds_off_bytes,
                                            const void* gptr, unsigned dsz_code,
                                            unsigned tile_w, unsigned tile_h,
                                            unsigned tensor_w, unsigned tensor_h,
                                            unsigned row_stride_elems) {
  unsigned long long ga = (unsigned long long)(uintptr_t)gptr;
  u32x4 g0;
  g0[0] = 1u;                                   // count=1, user descriptor
  g0[1] = lds_off_bytes;                        // lds_addr (bytes)
  g0[2] = (unsigned)(ga & 0xFFFFFFFFu);         // global_addr[31:0]
  g0[3] = (unsigned)((ga >> 32) & 0x1FFFFFFu)   // global_addr[56:32]
        | (2u << 30);                           // type=2 ("image")
  i32x8 g1;
  g1[0] = (int)(dsz_code << 16);                // workgroup_mask=0, data_size
  g1[1] = (int)((tensor_w & 0xFFFFu) << 16);    // tensor_dim0 lo16 @ bits 63:48
  g1[2] = (int)(((tensor_w >> 16) & 0xFFFFu) |
                ((tensor_h & 0xFFFFu) << 16));  // tensor_dim0 hi / dim1 lo
  g1[3] = (int)(((tensor_h >> 16) & 0xFFFFu) |
                ((tile_w & 0xFFFFu) << 16));    // tensor_dim1 hi / tile_dim0
  g1[4] = (int)(tile_h & 0xFFFFu);              // tile_dim1 (tile_dim2=0)
  g1[5] = (int)row_stride_elems;                // tensor_dim0_stride[31:0]
  g1[6] = 0;
  g1[7] = 0;
  i32x4 gz = {0, 0, 0, 0};
#if defined(__clang_major__) && (__clang_major__ >= 23)
  i32x8 gz8 = {0, 0, 0, 0, 0, 0, 0, 0};
  __builtin_amdgcn_tensor_load_to_lds(g0, g1, gz, gz, gz8, 0);
#else
  __builtin_amdgcn_tensor_load_to_lds(g0, g1, gz, gz, 0);
#endif
}

// ---- P0: one-time f32 -> bf16 conversion -----------------------------------
__global__ void cvt_f32_to_bf16(const float* __restrict__ src,
                                __bf16* __restrict__ dst, long n) {
  long i = (long)blockIdx.x * blockDim.x + threadIdx.x;
  long stride = (long)gridDim.x * blockDim.x;
  for (; i < n; i += stride) dst[i] = (__bf16)src[i];
}

// ---- K1: 32x64 output strip per wave (8 accumulators) ----------------------
// A reused x4 (across N-frags), B reused x2 (across M-frags). K % 32 == 0.
__global__ void gemm_bf16_wmma_32x64(const __bf16* __restrict__ A,
                                     const __bf16* __restrict__ Bw,
                                     float* __restrict__ C,
                                     int M, int N, int K, int lda, int ldb, int ldc) {
  const int lane = threadIdx.x & 31;
  const int wave = threadIdx.x >> 5;
  const int tile = blockIdx.x * (blockDim.x >> 5) + wave;
  const int ntn  = N >> 6;
  const int tm   = tile / ntn;         // 32-row strip
  const int tn   = tile - tm * ntn;    // 64-col strip
  if (tm * 32 >= M) return;

  const __bf16* arow0 = A + (size_t)(tm * 32 + (lane & 15)) * lda;
  const __bf16* arow1 = arow0 + (size_t)16 * lda;
  const __bf16* brow[4];
#pragma unroll
  for (int f = 0; f < 4; ++f)
    brow[f] = Bw + (size_t)(tn * 64 + f * 16 + (lane & 15)) * ldb;

  v8f acc[8];
#pragma unroll
  for (int i = 0; i < 8; ++i) acc[i] = (v8f){0.f,0.f,0.f,0.f,0.f,0.f,0.f,0.f};

  for (int k0 = 0; k0 < K; k0 += 32) {
    if (k0 + 32 < K) {
      __builtin_prefetch(arow0 + k0 + 32, 0, 3);   // global_prefetch_b8
      __builtin_prefetch(arow1 + k0 + 32, 0, 3);
    }
    v16bf a0 = load_a_bf(arow0, k0, lane);
    v16bf a1 = load_a_bf(arow1, k0, lane);
    v16bf b0 = load_b_bf(brow[0], k0, lane);
    v16bf b1 = load_b_bf(brow[1], k0, lane);
    v16bf b2 = load_b_bf(brow[2], k0, lane);
    v16bf b3 = load_b_bf(brow[3], k0, lane);
    acc[0] = __builtin_amdgcn_wmma_f32_16x16x32_bf16(false, a0, false, b0, (short)0, acc[0], false, false);
    acc[1] = __builtin_amdgcn_wmma_f32_16x16x32_bf16(false, a1, false, b0, (short)0, acc[1], false, false);
    acc[2] = __builtin_amdgcn_wmma_f32_16x16x32_bf16(false, a0, false, b1, (short)0, acc[2], false, false);
    acc[3] = __builtin_amdgcn_wmma_f32_16x16x32_bf16(false, a1, false, b1, (short)0, acc[3], false, false);
    acc[4] = __builtin_amdgcn_wmma_f32_16x16x32_bf16(false, a0, false, b2, (short)0, acc[4], false, false);
    acc[5] = __builtin_amdgcn_wmma_f32_16x16x32_bf16(false, a1, false, b2, (short)0, acc[5], false, false);
    acc[6] = __builtin_amdgcn_wmma_f32_16x16x32_bf16(false, a0, false, b3, (short)0, acc[6], false, false);
    acc[7] = __builtin_amdgcn_wmma_f32_16x16x32_bf16(false, a1, false, b3, (short)0, acc[7], false, false);
  }

  // D layout: VGPR j -> (lanes 0-15: M=j, N=lane) (lanes 16-31: M=8+j, N=lane-16)
  const int n  = lane & 15;
  const int mb = (lane >> 4) << 3;
#pragma unroll
  for (int f = 0; f < 4; ++f) {
    int cn = tn * 64 + f * 16 + n;
#pragma unroll
    for (int r = 0; r < 2; ++r) {
#pragma unroll
      for (int j = 0; j < 8; ++j) {
        int rm = tm * 32 + r * 16 + mb + j;
        C[(size_t)rm * ldc + cn] = acc[f * 2 + r][j];
      }
    }
  }
}

// ---- K3: 16x16 tile per wave (N=80 -> 5 tiles) -----------------------------
__global__ void gemm_bf16_wmma16(const __bf16* __restrict__ A,
                                 const __bf16* __restrict__ Bw,
                                 float* __restrict__ C,
                                 int M, int N, int K, int lda, int ldb, int ldc) {
  const int lane = threadIdx.x & 31;
  const int wave = threadIdx.x >> 5;
  const int tile = blockIdx.x * (blockDim.x >> 5) + wave;
  const int ntn  = N >> 4;
  const int tm   = tile / ntn;
  const int tn   = tile - tm * ntn;
  if (tm * 16 >= M) return;

  const __bf16* arow = A  + (size_t)(tm * 16 + (lane & 15)) * lda;
  const __bf16* brow = Bw + (size_t)(tn * 16 + (lane & 15)) * ldb;

  v8f acc = {0.f, 0.f, 0.f, 0.f, 0.f, 0.f, 0.f, 0.f};
  for (int k0 = 0; k0 < K; k0 += 32) {
    if (k0 + 32 < K) {
      __builtin_prefetch(arow + k0 + 32, 0, 3);
      __builtin_prefetch(brow + k0 + 32, 0, 3);
    }
    v16bf af = load_a_bf(arow, k0, lane);
    v16bf bf = load_b_bf(brow, k0, lane);
    acc = __builtin_amdgcn_wmma_f32_16x16x32_bf16(
        false, af, false, bf, (short)0, acc, false, false);
  }

  const int n  = lane & 15;
  const int mb = (lane >> 4) << 3;
  const int cn = tn * 16 + n;
#pragma unroll
  for (int j = 0; j < 8; ++j)
    C[(size_t)(tm * 16 + mb + j) * ldc + cn] = acc[j];
}

// ---- K4: delta = softplus(dbc[:, :48] @ dt_proj_w^T + b) -------------------
// Block = 4 waves, 64x64 tile. bf16 B tile (64 x 48, 6 KB) staged into LDS by
// the Tensor Data Mover; fragments then read from LDS (ds_load). K=48 guard
// collapses to whole-chunk selects (chunk starts are multiples of 8/16).
__global__ void gemm_dtproj_tdm(const float*  __restrict__ A,    // dbc   [ROWS, 80]
                                const __bf16* __restrict__ Bw,   // dt_w  bf16 [1536, 48]
                                const float*  __restrict__ bias, // dt_b  [1536]
                                float* __restrict__ C) {         // delta [ROWS, 1536]
  __shared__ __bf16 ldsB[64 * DTRANK];     // 6 KB of the 320 KB/WGP LDS
  const int lane = threadIdx.x & 31;
  const int wave = threadIdx.x >> 5;
  const int ntn  = DINNER / 64;            // 24
  const int bn   = blockIdx.x % ntn;
  const int bm   = blockIdx.x / ntn;

  if (threadIdx.x < 32) {                  // wave 0 issues the DMA
    tdm_load_2d((unsigned)(uintptr_t)&ldsB[0],
                Bw + (size_t)(bn * 64) * DTRANK, /*dsz=2B*/1u,
                /*tile_w=*/DTRANK, /*tile_h=*/64,
                /*tensor_w=*/DTRANK, /*tensor_h=*/DINNER,
                /*row_stride=*/DTRANK);
    __builtin_amdgcn_s_wait_tensorcnt(0);
  }
  __syncthreads();                         // tile visible to all 4 waves

  const int mrow = bm * 64 + wave * 16 + (lane & 15);
  const float* arow = A + (size_t)mrow * XPROJ_N;
  const int kb = (lane >> 4) << 3;
  const int kh = (lane >> 4) << 4;

  v8f acc[4];
#pragma unroll
  for (int i = 0; i < 4; ++i) acc[i] = (v8f){0.f,0.f,0.f,0.f,0.f,0.f,0.f,0.f};

#pragma unroll
  for (int k0 = 0; k0 < 64; k0 += 32) {    // Kpad(48) = 64
    v8bf alo = cvt_chunk8_guard(arow, k0 + kh, DTRANK);
    v8bf ahi = cvt_chunk8_guard(arow, k0 + kh + 8, DTRANK);
    v16bf af = __builtin_shufflevector(alo, ahi, 0,1,2,3,4,5,6,7,
                                                 8,9,10,11,12,13,14,15);
#pragma unroll
    for (int f = 0; f < 4; ++f) {
      const __bf16* lrow = ldsB + (f * 16 + (lane & 15)) * DTRANK;
      const int ks = k0 + kh;              // 0,16,32,48
      v16bf bf;
      if (ks < DTRANK) {
        bf = frag16(lrow + ks, lrow + ks + 8);   // ds_load from staged tile
      } else {
#pragma unroll
        for (int i = 0; i < 16; ++i) bf[i] = (__bf16)0.f;
      }
      acc[f] = __builtin_amdgcn_wmma_f32_16x16x32_bf16(
          false, af, false, bf, (short)0, acc[f], false, false);
    }
  }

  const int n  = lane & 15;
  const int mb = (lane >> 4) << 3;
#pragma unroll
  for (int f = 0; f < 4; ++f) {
    int cn = bn * 64 + f * 16 + n;
    float bv = bias[cn];
#pragma unroll
    for (int j = 0; j < 8; ++j) {
      int rm = bm * 64 + wave * 16 + mb + j;
      C[(size_t)rm * DINNER + cn] = softplus_f(acc[f][j] + bv);
    }
  }
}

// ---- K2: causal depthwise conv1d (width 4) + bias + SiLU (f32 + bf16 out) --
__global__ void conv_silu_kernel(const float* __restrict__ xz,     // [ROWS, 3072]
                                 const float* __restrict__ conv_w, // [1536, 4]
                                 const float* __restrict__ conv_b, // [1536]
                                 float* __restrict__ xi_c,         // [ROWS, 1536]
                                 __bf16* __restrict__ xi_b) {      // bf16 copy
  int idx = blockIdx.x * blockDim.x + threadIdx.x;
  if (idx >= ROWS * DINNER) return;
  int e  = idx % DINNER;
  int bl = idx / DINNER;
  int l  = bl % LL;
  float acc = conv_b[e];
  const float* w = conv_w + e * 4;
#pragma unroll
  for (int t = 0; t < 4; ++t) {
    int ls = l - 3 + t;
    if (ls >= 0) acc += w[t] * xz[(size_t)(bl - (3 - t)) * (2 * DINNER) + e];
  }
  float v = acc * sigmoid_f(acc);
  xi_c[idx] = v;
  xi_b[idx] = (__bf16)v;
}

// ---- K5: selective scan, one thread per (b,e); fused gate/skip/mean --------
__global__ void scan_kernel(const float* __restrict__ xz,     // z = xz[:, 1536:]
                            const float* __restrict__ xi_c,   // [ROWS, 1536]
                            const float* __restrict__ dbc,    // [ROWS, 80]
                            const float* __restrict__ delta,  // [ROWS, 1536]
                            const float* __restrict__ A_log,  // [1536, 16]
                            const float* __restrict__ Dvec,   // [1536]
                            float* __restrict__ ybar) {       // [8, 1536]
  int tid = blockIdx.x * blockDim.x + threadIdx.x;
  if (tid >= BB * DINNER) return;
  int e = tid % DINNER;
  int b = tid / DINNER;

  float Ae[DSTATE], h[DSTATE];
#pragma unroll
  for (int n = 0; n < DSTATE; ++n) {
    Ae[n] = -__expf(A_log[e * DSTATE + n]);
    h[n]  = 0.f;
  }
  const float Dv = Dvec[e];
  float acc = 0.f;

  for (int l = 0; l < LL; ++l) {
    size_t idx = (size_t)b * LL + l;
    float dlt = delta[idx * DINNER + e];
    float xt  = xi_c[idx * DINNER + e];
    float zt  = xz[idx * (2 * DINNER) + DINNER + e];
    const float* bc = dbc + idx * XPROJ_N;   // wave-uniform (broadcast) loads
    float dx = dlt * xt;
    float yt = 0.f;
#pragma unroll
    for (int n = 0; n < DSTATE; ++n) {
      float dA = __expf(dlt * Ae[n]);
      h[n] = dA * h[n] + dx * bc[DTRANK + n];
      yt  += h[n] * bc[DTRANK + DSTATE + n];
    }
    float y = yt + xt * Dv;
    acc += y * (zt * sigmoid_f(zt));
  }
  ybar[tid] = acc * (1.f / (float)LL);
}

// ---- K6: fold out_proj + fc: M[2,1536] = fc_w[2,768] @ out_proj_w ---------
__global__ void fold_w_kernel(const float* __restrict__ fc_w,
                              const float* __restrict__ opw,
                              float* __restrict__ Mf) {
  int idx = blockIdx.x * blockDim.x + threadIdx.x;
  if (idx >= 2 * DINNER) return;
  int e = idx % DINNER;
  int o = idx / DINNER;
  float s = 0.f;
  for (int d = 0; d < DMODEL; ++d)
    s += fc_w[o * DMODEL + d] * opw[(size_t)d * DINNER + e];
  Mf[idx] = s;
}

// ---- K7: out[b,o] = sum_e ybar[b,e] * M[o,e] + fc_b[o] --------------------
__global__ void final_kernel(const float* __restrict__ ybar,
                             const float* __restrict__ Mf,
                             const float* __restrict__ fc_b,
                             float* __restrict__ out) {
  __shared__ float red[256];
  int o = blockIdx.x & 1;
  int b = blockIdx.x >> 1;
  float s = 0.f;
  for (int e = threadIdx.x; e < DINNER; e += blockDim.x)
    s += ybar[b * DINNER + e] * Mf[o * DINNER + e];
  red[threadIdx.x] = s;
  __syncthreads();
  for (int st = 128; st > 0; st >>= 1) {
    if ((int)threadIdx.x < st) red[threadIdx.x] += red[threadIdx.x + st];
    __syncthreads();
  }
  if (threadIdx.x == 0) out[b * 2 + o] = red[0] + fc_b[o];
}

// ---------------------------------------------------------------------------
extern "C" void kernel_launch(void* const* d_in, const int* in_sizes, int n_in,
                              void* d_out, int out_size, void* d_ws, size_t ws_size,
                              hipStream_t stream) {
  const float* x         = (const float*)d_in[0];   // [8,2048,768]
  const float* in_proj_w = (const float*)d_in[1];   // [3072,768]
  const float* conv_w    = (const float*)d_in[2];   // [1536,4]
  const float* conv_b    = (const float*)d_in[3];   // [1536]
  const float* x_proj_w  = (const float*)d_in[4];   // [80,1536]
  const float* dt_proj_w = (const float*)d_in[5];   // [1536,48]
  const float* dt_proj_b = (const float*)d_in[6];   // [1536]
  const float* A_log     = (const float*)d_in[7];   // [1536,16]
  const float* Dvec      = (const float*)d_in[8];   // [1536]
  const float* opw       = (const float*)d_in[9];   // [768,1536]
  const float* fc_w      = (const float*)d_in[10];  // [2,768]
  const float* fc_b      = (const float*)d_in[11];  // [2]
  float* out = (float*)d_out;                       // [8,2] f32

  // workspace layout (256B-aligned sections)
  char*  wsc = (char*)d_ws;
  size_t cur = 0;
  auto take = [&](size_t bytes) -> void* {
    void* p = wsc + cur;
    cur += (bytes + 255) & ~(size_t)255;
    return p;
  };
  float*  xz    = (float*) take((size_t)ROWS * 2 * DINNER * 4);  // 201 MB
  float*  xi_c  = (float*) take((size_t)ROWS * DINNER * 4);      // 101 MB
  float*  dbc   = (float*) take((size_t)ROWS * XPROJ_N * 4);     //   5 MB
  float*  delta = (float*) take((size_t)ROWS * DINNER * 4);      // 101 MB
  float*  ybar  = (float*) take((size_t)BB * DINNER * 4);
  float*  Mf    = (float*) take((size_t)2 * DINNER * 4);
  __bf16* xb    = (__bf16*)take((size_t)ROWS * DMODEL * 2);      //  25 MB
  __bf16* wb_in = (__bf16*)take((size_t)2 * DINNER * DMODEL * 2);//   5 MB
  __bf16* wb_xp = (__bf16*)take((size_t)XPROJ_N * DINNER * 2);
  __bf16* wb_dt = (__bf16*)take((size_t)DINNER * DTRANK * 2);
  __bf16* xi_b  = (__bf16*)take((size_t)ROWS * DINNER * 2);      //  50 MB

  const int TPB = 128;                 // 4 waves per block
  const int WPB = TPB / 32;

  // P0: one-time bf16 conversions of GEMM operands
  {
    long n0 = (long)ROWS * DMODEL;
    cvt_f32_to_bf16<<<(int)((n0 + 1023) / 1024), 1024, 0, stream>>>(x, xb, n0);
    long n1 = (long)2 * DINNER * DMODEL;
    cvt_f32_to_bf16<<<(int)((n1 + 1023) / 1024), 1024, 0, stream>>>(in_proj_w, wb_in, n1);
    long n2 = (long)XPROJ_N * DINNER;
    cvt_f32_to_bf16<<<(int)((n2 + 1023) / 1024), 1024, 0, stream>>>(x_proj_w, wb_xp, n2);
    long n3 = (long)DINNER * DTRANK;
    cvt_f32_to_bf16<<<(int)((n3 + 1023) / 1024), 1024, 0, stream>>>(dt_proj_w, wb_dt, n3);
  }
  // K1: xz = x @ in_proj_w^T  (M=16384, N=3072, K=768), 32x64 per wave
  {
    int tiles = (ROWS / 32) * ((2 * DINNER) / 64);   // 512*48 = 24576
    gemm_bf16_wmma_32x64<<<tiles / WPB, TPB, 0, stream>>>(
        xb, wb_in, xz, ROWS, 2 * DINNER, DMODEL, DMODEL, DMODEL, 2 * DINNER);
  }
  // K2: causal conv + SiLU (emits f32 for scan + bf16 for K3)
  {
    int total = ROWS * DINNER;
    conv_silu_kernel<<<(total + 255) / 256, 256, 0, stream>>>(
        xz, conv_w, conv_b, xi_c, xi_b);
  }
  // K3: dbc = xi_c @ x_proj_w^T  (M=16384, N=80, K=1536), 16x16 per wave
  {
    int tiles = (ROWS / 16) * (XPROJ_N / 16);        // 5120
    gemm_bf16_wmma16<<<tiles / WPB, TPB, 0, stream>>>(
        xi_b, wb_xp, dbc, ROWS, XPROJ_N, DINNER, DINNER, DINNER, XPROJ_N);
  }
  // K4: delta = softplus(dbc[:, :48] @ dt_proj_w^T + b), TDM-staged bf16 B
  {
    int blocks = (ROWS / 64) * (DINNER / 64);        // 256*24 = 6144
    gemm_dtproj_tdm<<<blocks, TPB, 0, stream>>>(dbc, wb_dt, dt_proj_b, delta);
  }
  // K5: selective scan + gate + D-skip + time mean
  {
    int total = BB * DINNER;                          // 12288 threads
    scan_kernel<<<(total + 255) / 256, 256, 0, stream>>>(
        xz, xi_c, dbc, delta, A_log, Dvec, ybar);
  }
  // K6: fold fc_w @ out_proj_w
  fold_w_kernel<<<(2 * DINNER + 255) / 256, 256, 0, stream>>>(fc_w, opw, Mf);
  // K7: final projection
  final_kernel<<<BB * 2, 256, 0, stream>>>(ybar, Mf, fc_b, out);

  (void)in_sizes; (void)n_in; (void)out_size; (void)ws_size;
}
